// customLSTM_42434276884625
// MI455X (gfx1250) — matrix operand order repaired and checked by reading, not compiled, not run on hardware
//
#include <hip/hip_runtime.h>
#include <hip/hip_bf16.h>
#include <stdint.h>

// ---- Problem constants (match reference) ----
#define B_  64
#define T_  512
#define D_  1024
#define H_  1024
#define G_  4096   // 4*H

typedef __attribute__((ext_vector_type(16))) __bf16 v16bf;
typedef __attribute__((ext_vector_type(8)))  float  v8f;
typedef __attribute__((ext_vector_type(4)))  int    v4i;   // native clang vector (builtin param type)

union Frag32B { v16bf v; uint4 q[2]; };

#if defined(__gfx1250__) && __has_builtin(__builtin_amdgcn_sched_barrier)
#define SCHED_BARRIER() __builtin_amdgcn_sched_barrier(0)
#else
#define SCHED_BARRIER()
#endif

// Async global->LDS path (CDNA5 GLOBAL_LOAD_ASYNC_TO_LDS_B128, ASYNCcnt)
#if defined(__gfx1250__) && __has_builtin(__builtin_amdgcn_global_load_async_to_lds_b128)
#define USE_ASYNC_LDS 1
#define ASG __attribute__((address_space(1)))
#define ASL __attribute__((address_space(3)))
#if __has_builtin(__builtin_amdgcn_s_wait_asynccnt)
#define WAIT_ASYNC(n) __builtin_amdgcn_s_wait_asynccnt(n)
#else
#define WAIT_ASYNC(n) asm volatile("s_wait_asynccnt %0" :: "i"(n) : "memory")
#endif
#else
#define USE_ASYNC_LDS 0
#endif

// ---------------------------------------------------------------------------
// A-fragment (16x32 bf16, row-major source). ISA 7.12.2:
// lanes 0-15 : M=lane,   elems 0-7 -> K=k0..k0+7,   elems 8-15 -> K=k0+16..k0+23
// lanes 16-31: M=lane-16,elems 0-7 -> K=k0+8..k0+15,elems 8-15 -> K=k0+24..k0+31
// ---------------------------------------------------------------------------
__device__ __forceinline__ v16bf load_a_rowmajor(const __bf16* row, int k0, int lane) {
    const int half = lane >> 4;
    Frag32B a;
    a.q[0] = *(const uint4*)(row + k0 + 8 * half);
    a.q[1] = *(const uint4*)(row + k0 + 16 + 8 * half);
    return a.v;
}

// B-fragment from pre-swizzled storage: tile (ntile,ktile) of 32(K)x16(N),
// 512 bf16 stored lane-major (lane*16 contiguous elems in VGPR order).
__device__ __forceinline__ v16bf load_b_swz(const __bf16* swz, int ktiles,
                                            int ntile, int ktile, int lane) {
    const __bf16* p = swz + ((size_t)(ntile * ktiles + ktile) << 9) + lane * 16;
    Frag32B b;
    b.q[0] = ((const uint4*)p)[0];
    b.q[1] = ((const uint4*)p)[1];
    return b.v;
}

__device__ __forceinline__ float hsig(float z) {
    return fminf(fmaxf(0.2f * z + 0.5f, 0.0f), 1.0f);
}

// ---------------------------------------------------------------------------
// Phase 1 kernels: precision conversion / layout building
// ---------------------------------------------------------------------------
__global__ void k_convert_x(const float* __restrict__ x, __bf16* __restrict__ xbf, int n) {
    int i = blockIdx.x * 256 + threadIdx.x;
    if (i < n) xbf[i] = (__bf16)x[i];
}

// Build concatenated [K, 4H] bf16 weight in WMMA-B swizzled tile layout.
// Element j of lane l in tile (ntile,ktile) is
//   B(k = ktile*32 + 8*(l>>4) + (j&7) + 16*(j>>3),  n = ntile*16 + (l&15))
__global__ void k_build_swz(const float* __restrict__ W0, const float* __restrict__ W1,
                            const float* __restrict__ W2, const float* __restrict__ W3,
                            __bf16* __restrict__ out, int K) {
    size_t o = (size_t)blockIdx.x * 256 + threadIdx.x;
    if (o >= (size_t)K * G_) return;
    const int ktiles = K >> 5;
    size_t tile = o >> 9;
    int r = (int)(o & 511);
    int l = r >> 4, j = r & 15;
    int ntile = (int)(tile / ktiles);
    int ktile = (int)(tile % ktiles);
    int k = ktile * 32 + ((l >> 4) << 3) + (j & 7) + ((j >> 3) << 4);
    int n = (ntile << 4) + (l & 15);
    int gate = n >> 10, col = n & 1023;
    const float* Wg = (gate == 0) ? W0 : (gate == 1) ? W1 : (gate == 2) ? W2 : W3;
    out[o] = (__bf16)Wg[(size_t)k * H_ + col];
}

__global__ void k_build_bias(const float* __restrict__ b0, const float* __restrict__ b1,
                             const float* __restrict__ b2, const float* __restrict__ b3,
                             float* __restrict__ bx) {
    int n = blockIdx.x * 256 + threadIdx.x;
    if (n >= G_) return;
    int gate = n >> 10, col = n & 1023;
    const float* bg = (gate == 0) ? b0 : (gate == 1) ? b1 : (gate == 2) ? b2 : b3;
    bx[n] = bg[col];
}

__global__ void k_init_state(const float* __restrict__ c0, const float* __restrict__ h0,
                             float* __restrict__ cbuf, __bf16* __restrict__ hbf0) {
    int i = blockIdx.x * 256 + threadIdx.x;
    if (i >= B_ * H_) return;
    cbuf[i] = c0[i];
    hbf0[i] = (__bf16)h0[i];
}

// ---------------------------------------------------------------------------
// Phase 2: xp[t*B + b, :] = x[b,t,:] @ Wx + bx     (M=32768, N=4096, K=1024)
// Workgroup = 4 waves, 128(M) x 64(N); wave = 32(M) x 64(N): 2 A frags,
// 8 accumulators, register double-buffered K pipeline enforced by
// sched_barrier so loads for kt+1 stay in flight across the wmmas of kt.
// ---------------------------------------------------------------------------
__global__ __launch_bounds__(128) void k_gemm_xp(const __bf16* __restrict__ xbf,
                                                 const __bf16* __restrict__ wx,
                                                 const float* __restrict__ bx,
                                                 float* __restrict__ xp) {
    const int wave = threadIdx.x >> 5;
    const int lane = threadIdx.x & 31;
    const int nblk = blockIdx.x & 63;    // 64 N-blocks of 64 cols
    const int mblk = blockIdx.x >> 6;    // 256 M-blocks of 128 rows
    const int m0 = mblk * 128 + wave * 32;
    const int n0 = nblk * 64;
    const int KT = D_ >> 5;              // 32 k-tiles

    const int mA0 = m0 + (lane & 15);
    const int mA1 = m0 + 16 + (lane & 15);
    const __bf16* arow0 = xbf + ((size_t)(mA0 & 63) * T_ + (mA0 >> 6)) * D_;
    const __bf16* arow1 = xbf + ((size_t)(mA1 & 63) * T_ + (mA1 >> 6)) * D_;

    const int ncol = lane & 15;
    v8f acc[2][4];
#pragma unroll
    for (int s = 0; s < 4; ++s) {
        float bv = bx[n0 + s * 16 + ncol];
#pragma unroll
        for (int r = 0; r < 8; ++r) { acc[0][s][r] = bv; acc[1][s][r] = bv; }
    }

    v16bf abuf[2][2];
    v16bf bbuf[2][4];
    abuf[0][0] = load_a_rowmajor(arow0, 0, lane);
    abuf[0][1] = load_a_rowmajor(arow1, 0, lane);
#pragma unroll
    for (int s = 0; s < 4; ++s)
        bbuf[0][s] = load_b_swz(wx, KT, (n0 >> 4) + s, 0, lane);

#pragma unroll
    for (int kt = 0; kt < 32; ++kt) {
        const int cur = kt & 1, nxt = cur ^ 1;
        if (kt + 1 < 32) {
            abuf[nxt][0] = load_a_rowmajor(arow0, (kt + 1) << 5, lane);
            abuf[nxt][1] = load_a_rowmajor(arow1, (kt + 1) << 5, lane);
#pragma unroll
            for (int s = 0; s < 4; ++s)
                bbuf[nxt][s] = load_b_swz(wx, KT, (n0 >> 4) + s, kt + 1, lane);
        }
        if (kt + 4 < 32) {
            __builtin_prefetch(arow0 + ((kt + 4) << 5), 0, 1);
            __builtin_prefetch(arow1 + ((kt + 4) << 5), 0, 1);
        }
        SCHED_BARRIER();   // loads above may not sink past, wmmas may not hoist
#pragma unroll
        for (int s = 0; s < 4; ++s) {
            acc[0][s] = __builtin_amdgcn_wmma_f32_16x16x32_bf16(
                false, abuf[cur][0], false, bbuf[cur][s], (short)0, acc[0][s], false, false);
            acc[1][s] = __builtin_amdgcn_wmma_f32_16x16x32_bf16(
                false, abuf[cur][1], false, bbuf[cur][s], (short)0, acc[1][s], false, false);
        }
        SCHED_BARRIER();
    }

    const int rbase = (lane >> 4) << 3;
#pragma unroll
    for (int h = 0; h < 2; ++h)
#pragma unroll
        for (int s = 0; s < 4; ++s)
#pragma unroll
            for (int r = 0; r < 8; ++r)
                xp[(size_t)(m0 + h * 16 + rbase + r) * G_ + n0 + s * 16 + ncol] = acc[h][s][r];
}

// ---------------------------------------------------------------------------
// Recurrent step (one launch per t):
//   g = xp[t] + h_bf @ Wh  for gate columns {j0.., H+j0.., 2H+j0.., 3H+j0..}
//   then fused gate nonlinearity + c/h update in registers.
// Gate-phase pipeline: while gate s is multiplied (32 wmma), gate s+1's
// 32 KB B-slice is async-copied global->LDS (ASYNCcnt); B frags then come
// from LDS (ds_load_b128, short latency). Falls back to direct global B
// loads when the async builtin is unavailable.
// ---------------------------------------------------------------------------
__global__ __launch_bounds__(128) void k_step(const __bf16* __restrict__ hbf_in,
                                              __bf16* __restrict__ hbf_out,
                                              const __bf16* __restrict__ wh,
                                              const float* __restrict__ xp_t,
                                              float* __restrict__ cbuf,
                                              float* __restrict__ out_h,
                                              int t) {
    const int wave = threadIdx.x >> 5;
    const int lane = threadIdx.x & 31;
    const int j0 = blockIdx.x << 4;      // 16 H-columns per block
    const int m0 = wave << 4;            // 16 batch rows per wave
    const int ncol = lane & 15;
    const int rbase = (lane >> 4) << 3;
    const int KT = H_ >> 5;              // 32 k-tiles

    const __bf16* arow = hbf_in + (size_t)(m0 + (lane & 15)) * H_;

#if USE_ASYNC_LDS
    extern __shared__ __align__(16) __bf16 ldsb[];   // 4 gates * KT*512 bf16 = 128 KB
    auto stage_gate = [&](int s) {
        const __bf16* gsrc = wh + (size_t)(s * (H_ >> 4) + (j0 >> 4)) * KT * 512;
        __bf16* ldst = ldsb + (size_t)s * KT * 512;
#pragma unroll
        for (int c = 0; c < 16; ++c) {           // 16 x 16B per thread = 32 KB/WG
            const int e = (c * 128 + (int)threadIdx.x) * 8;
            __builtin_amdgcn_global_load_async_to_lds_b128(
                (ASG v4i*)(uintptr_t)(gsrc + e),
                (ASL v4i*)(uint32_t)(uintptr_t)(ldst + e),
                0, 0);
        }
    };
    stage_gate(0);
#endif

    // acc[s] accumulates gate s (f,i,o,c) at columns j0..j0+15, init from xp_t
    v8f acc[4];
#pragma unroll
    for (int s = 0; s < 4; ++s)
#pragma unroll
        for (int r = 0; r < 8; ++r)
            acc[s][r] = xp_t[(size_t)(m0 + rbase + r) * G_ + s * H_ + j0 + ncol];

    auto loadB = [&](int s, int kt) -> v16bf {
#if USE_ASYNC_LDS
        const __bf16* p = ldsb + (((size_t)s * KT + kt) << 9) + lane * 16;
        Frag32B b;
        b.q[0] = ((const uint4*)p)[0];
        b.q[1] = ((const uint4*)p)[1];
        return b.v;
#else
        return load_b_swz(wh, KT, s * (H_ >> 4) + (j0 >> 4), kt, lane);
#endif
    };

#pragma unroll
    for (int s = 0; s < 4; ++s) {
#if USE_ASYNC_LDS
        if (s + 1 < 4) stage_gate(s + 1);
        // In-order ASYNCcnt: <=16 outstanding  ==>  gate s fully in LDS.
        if (s < 3) WAIT_ASYNC(16); else WAIT_ASYNC(0);
        __syncthreads();
#endif
        v16bf ab[2], bb[2];
        ab[0] = load_a_rowmajor(arow, 0, lane);
        bb[0] = loadB(s, 0);
#pragma unroll
        for (int kt = 0; kt < 32; ++kt) {
            const int cur = kt & 1, nxt = cur ^ 1;
            if (kt + 1 < 32) {
                ab[nxt] = load_a_rowmajor(arow, (kt + 1) << 5, lane);
                bb[nxt] = loadB(s, kt + 1);
            }
            SCHED_BARRIER();
            acc[s] = __builtin_amdgcn_wmma_f32_16x16x32_bf16(
                false, ab[cur], false, bb[cur], (short)0, acc[s], false, false);
            SCHED_BARRIER();
        }
    }

    // Fused LSTM pointwise update on the 8 (m, j) elements this lane owns.
#pragma unroll
    for (int r = 0; r < 8; ++r) {
        const int mm = m0 + rbase + r;       // batch index
        const int j  = j0 + ncol;            // hidden index
        const float f = hsig(acc[0][r]);
        const float i = hsig(acc[1][r]);
        const float o = hsig(acc[2][r]);
        const float ct = tanhf(acc[3][r]);
        const size_t cidx = (size_t)mm * H_ + j;
        const float cn = f * cbuf[cidx] + i * ct;
        const float h = o * tanhf(cn);
        cbuf[cidx] = cn;
        out_h[((size_t)mm * T_ + t) * H_ + j] = h;   // output [B,T,H]
        hbf_out[cidx] = (__bf16)h;
    }
}

// ---------------------------------------------------------------------------
// Launch
// ---------------------------------------------------------------------------
extern "C" void kernel_launch(void* const* d_in, const int* in_sizes, int n_in,
                              void* d_out, int out_size, void* d_ws, size_t ws_size,
                              hipStream_t stream) {
    const float* x   = (const float*)d_in[0];
    const float* c0  = (const float*)d_in[1];
    const float* h0  = (const float*)d_in[2];
    const float* Wfx = (const float*)d_in[3];
    const float* bf_ = (const float*)d_in[4];
    const float* Wfh = (const float*)d_in[5];
    const float* Wix = (const float*)d_in[6];
    const float* bi_ = (const float*)d_in[7];
    const float* Wih = (const float*)d_in[8];
    const float* Wox = (const float*)d_in[9];
    const float* bo_ = (const float*)d_in[10];
    const float* Woh = (const float*)d_in[11];
    const float* Wcx = (const float*)d_in[12];
    const float* bc_ = (const float*)d_in[13];
    const float* Wch = (const float*)d_in[14];
    float* out = (float*)d_out;

    // Workspace layout (all offsets 256B-aligned)
    char* ws = (char*)d_ws;
    size_t off = 0;
    __bf16* xbf = (__bf16*)(ws + off); off += (size_t)B_ * T_ * D_ * 2;   // 64 MB
    __bf16* wx  = (__bf16*)(ws + off); off += (size_t)D_ * G_ * 2;        // 8 MB
    __bf16* wh  = (__bf16*)(ws + off); off += (size_t)H_ * G_ * 2;        // 8 MB
    float*  bx  = (float*) (ws + off); off += (size_t)G_ * 4;             // 16 KB
    float*  xp  = (float*) (ws + off); off += (size_t)T_ * B_ * G_ * 4;   // 512 MB
    float*  cb  = (float*) (ws + off); off += (size_t)B_ * H_ * 4;        // 256 KB
    __bf16* hbf = (__bf16*)(ws + off); off += (size_t)2 * B_ * H_ * 2;    // 2x128 KB

    // Phase 1: conversions / layout
    {
        const int n = B_ * T_ * D_;
        k_convert_x<<<(n + 255) / 256, 256, 0, stream>>>(x, xbf, n);
    }
    k_build_swz<<<((size_t)D_ * G_ + 255) / 256, 256, 0, stream>>>(Wfx, Wix, Wox, Wcx, wx, D_);
    k_build_swz<<<((size_t)H_ * G_ + 255) / 256, 256, 0, stream>>>(Wfh, Wih, Woh, Wch, wh, H_);
    k_build_bias<<<(G_ + 255) / 256, 256, 0, stream>>>(bf_, bi_, bo_, bc_, bx);
    k_init_state<<<(B_ * H_ + 255) / 256, 256, 0, stream>>>(c0, h0, cb, hbf);

    // Phase 2: big input-projection GEMM (time-major xp)
    k_gemm_xp<<<(32768 / 128) * (G_ / 64), 128, 0, stream>>>(xbf, wx, bx, xp);

    // Phase 3: recurrence, one fused kernel per timestep (double-buffered h).
    // 128 KB dynamic LDS for the async-staged Wh slice (harmless if unused).
    const size_t step_lds = (size_t)4 * (H_ / 32) * 512 * sizeof(__bf16);
    for (int t = 0; t < T_; ++t) {
        const __bf16* hin = hbf + (size_t)(t & 1) * B_ * H_;
        __bf16* hout      = hbf + (size_t)((t + 1) & 1) * B_ * H_;
        k_step<<<H_ / 16, 128, step_lds, stream>>>(hin, hout, wh,
                                                   xp + (size_t)t * B_ * G_,
                                                   cb, out, t);
    }
}
